// HybridSTPAttention_36472862277839
// MI455X (gfx1250) — matrix operand order repaired and checked by reading, not compiled, not run on hardware
//
#include <hip/hip_runtime.h>
#include <hip/hip_bf16.h>
#include <math.h>

typedef _Float16 v16h __attribute__((ext_vector_type(16)));
typedef _Float16 v8h  __attribute__((ext_vector_type(8)));
typedef float    v8f  __attribute__((ext_vector_type(8)));

#define B_  2
#define L_  2048
#define D_  1024
#define H_  16
#define DH  64
#define BH  (B_*H_)
#define CH  128            // STP chunk length
#define NC  (L_/CH)        // 16 chunks

__device__ __forceinline__ float gelu_exact(float x) {
  return 0.5f * x * (1.0f + erff(x * 0.70710678118654752f));
}
__device__ __forceinline__ float sigmoidf_(float x) {
  return 1.0f / (1.0f + expf(-x));
}
// two contiguous 16B loads -> one 16-half fragment register pair
__device__ __forceinline__ v16h ld16(const _Float16* p0, const _Float16* p1) {
  v8h lo = *(const v8h*)p0;
  v8h hi = *(const v8h*)p1;
  return __builtin_shufflevector(lo, hi, 0,1,2,3,4,5,6,7,8,9,10,11,12,13,14,15);
}
// A-frag from row-major f16 (row length ldk): runs at k0+hs*8 and k0+16+hs*8
__device__ __forceinline__ v16h ldA(const _Float16* row, int k0, int hs) {
  return ld16(row + k0 + hs * 8, row + k0 + 16 + hs * 8);
}
// B-frag from transposed f16 (col-major: n-row of length ldk): runs at k0+hs*8 and k0+16+hs*8
__device__ __forceinline__ v16h ldB(const _Float16* nrow, int k0, int hs) {
  return ld16(nrow + k0 + hs * 8, nrow + k0 + 16 + hs * 8);
}

// ---------------------------------------------------------------------------
// Pack helpers: f32 -> f16 copy, and f32 (KxN) -> f16 transposed (NxK)
// ---------------------------------------------------------------------------
__global__ void cvt_f16(const float* __restrict__ in, _Float16* __restrict__ out, int n) {
  int i = blockIdx.x * blockDim.x + threadIdx.x;
  if (i < n) out[i] = (_Float16)in[i];
}
__global__ void transpose_cvt(const float* __restrict__ in, _Float16* __restrict__ out,
                              int K, int N) {
  int id = blockIdx.x * blockDim.x + threadIdx.x;
  if (id >= K * N) return;
  int k = id / N, n = id % N;
  out[(size_t)n * K + k] = (_Float16)in[id];
}

// ---------------------------------------------------------------------------
// Generic GEMM: A16 (MxK f16 row-major), BT16 (NxK f16, i.e. B transposed),
// f32 accumulate, f32 out. One 16x64 strip per wave (A-frag reused 4x).
// act: 0 = bias, 1 = bias + exact gelu
// ---------------------------------------------------------------------------
__global__ void gemm_f16wmma(const _Float16* __restrict__ A16, const _Float16* __restrict__ BT16,
                             const float* __restrict__ bias, float* __restrict__ C,
                             int M, int N, int K, int act) {
  const int lane = threadIdx.x & 31;
  const int wv   = threadIdx.x >> 5;
  const int nstrip = N >> 6;
  const int sid  = blockIdx.x * (blockDim.x >> 5) + wv;
  if (sid >= (M >> 4) * nstrip) return;
  const int tm  = (sid / nstrip) << 4;
  const int tn0 = (sid % nstrip) << 6;
  const int m = lane & 15, hs = lane >> 4;

  const _Float16* arow = A16 + (size_t)(tm + m) * K;
  v8f c[4] = {{}, {}, {}, {}};
  for (int k0 = 0; k0 < K; k0 += 32) {
    v16h a = ldA(arow, k0, hs);
#pragma unroll
    for (int nt = 0; nt < 4; ++nt) {
      v16h b = ldB(BT16 + (size_t)(tn0 + nt * 16 + m) * K, k0, hs);
      c[nt] = __builtin_amdgcn_wmma_f32_16x16x32_f16(false, a, false, b, (short)0, c[nt], false, false);
    }
  }
#pragma unroll
  for (int nt = 0; nt < 4; ++nt) {
#pragma unroll
    for (int r = 0; r < 8; ++r) {
      int row = tm + r + (hs << 3);
      int col = tn0 + nt * 16 + m;
      float v = c[nt][r] + bias[col];
      if (act == 1) v = gelu_exact(v);
      C[(size_t)row * N + col] = v;
    }
  }
}

// ---------------------------------------------------------------------------
// QKV GEMM (4096 x 3072, K=1024), 16x64 strip per wave.
// Epilogue packs (type,b,h,l,feat): f32 (q,k,v row-major) for STP, and f16 for
// attention with q,k row-major but v TRANSPOSED (bh,feat,L) so P@V B-frags are
// contiguous. k is pre-scaled by 1/sqrt(d) for both branches.
// ---------------------------------------------------------------------------
__global__ void qkv_gemm(const _Float16* __restrict__ x16, const _Float16* __restrict__ WT16,
                         const float* __restrict__ bias,
                         float* __restrict__ pk32, _Float16* __restrict__ pk16) {
  const int lane = threadIdx.x & 31;
  const int wv   = threadIdx.x >> 5;
  const int nstrip = 3072 >> 6;               // 48
  const int sid  = blockIdx.x * 8 + wv;
  const int tm  = (sid / nstrip) << 4;
  const int tn0 = (sid % nstrip) << 6;
  const int m = lane & 15, hs = lane >> 4;

  const _Float16* arow = x16 + (size_t)(tm + m) * 1024;
  v8f c[4] = {{}, {}, {}, {}};
  for (int k0 = 0; k0 < 1024; k0 += 32) {
    v16h a = ldA(arow, k0, hs);
#pragma unroll
    for (int nt = 0; nt < 4; ++nt) {
      v16h b = ldB(WT16 + (size_t)(tn0 + nt * 16 + m) * 1024, k0, hs);
      c[nt] = __builtin_amdgcn_wmma_f32_16x16x32_f16(false, a, false, b, (short)0, c[nt], false, false);
    }
  }
#pragma unroll
  for (int nt = 0; nt < 4; ++nt) {
#pragma unroll
    for (int r = 0; r < 8; ++r) {
      int token = tm + r + (hs << 3);         // b*L + l
      int n     = tn0 + nt * 16 + m;          // column in 3D
      int type  = n >> 10;
      int hh    = (n >> 6) & 15;
      int feat  = n & 63;
      int b     = token >> 11;
      int l     = token & 2047;
      int bh    = b * H_ + hh;
      float v = c[nt][r] + bias[n];
      if (type == 1) v *= 0.125f;             // k * 1/sqrt(d)
      size_t idx32 = ((size_t)(type * BH + bh) * L_ + l) * DH + feat;
      pk32[idx32] = v;
      size_t idx16 = (type == 2)
          ? ((size_t)(2 * BH + bh) * DH + feat) * L_ + l   // v transposed
          : idx32;                                         // q,k row-major
      pk16[idx16] = (_Float16)v;
    }
  }
}

// ---------------------------------------------------------------------------
// Flash attention, causal, per (b,h, 64-row q block). 4 waves x 16 q rows.
// ---------------------------------------------------------------------------
__global__ void attn_kernel(const _Float16* __restrict__ qkv16, float* __restrict__ ysm) {
  const int wv = threadIdx.x >> 5, lane = threadIdx.x & 31;
  const int m  = lane & 15, hs = lane >> 4;
  const int qb = blockIdx.x % (L_ / 64);
  const int bh = blockIdx.x / (L_ / 64);
  const int qr = qb * 64 + wv * 16;
  const _Float16* qh = qkv16 + ((size_t)(0 * BH + bh) * L_) * DH;   // (L, d)
  const _Float16* kh = qkv16 + ((size_t)(1 * BH + bh) * L_) * DH;   // (L, d)
  const _Float16* vT = qkv16 + ((size_t)(2 * BH + bh) * DH) * L_;   // (d, L)
  __shared__ _Float16 pls[4][16][32];

  const _Float16* qrow = qh + (size_t)(qr + m) * DH;
  v16h aq0 = ldA(qrow, 0, hs);
  v16h aq1 = ldA(qrow, 32, hs);

  v8f oacc[4] = {{}, {}, {}, {}};
  float mrow[8], lrow[8];
#pragma unroll
  for (int r = 0; r < 8; ++r) { mrow[r] = -1e30f; lrow[r] = 0.0f; }

  const int nkb = qb * 2 + 2;                 // same trip count for all 4 waves
  for (int kb = 0; kb < nkb; ++kb) {
    const int kc = kb * 32;
    v8f s0 = {}, s1 = {};
    {
      const _Float16* k0row = kh + (size_t)(kc + m) * DH;
      const _Float16* k1row = kh + (size_t)(kc + 16 + m) * DH;
      v16h b00 = ldB(k0row, 0, hs),  b01 = ldB(k0row, 32, hs);
      v16h b10 = ldB(k1row, 0, hs),  b11 = ldB(k1row, 32, hs);
      s0 = __builtin_amdgcn_wmma_f32_16x16x32_f16(false, aq0, false, b00, (short)0, s0, false, false);
      s0 = __builtin_amdgcn_wmma_f32_16x16x32_f16(false, aq1, false, b01, (short)0, s0, false, false);
      s1 = __builtin_amdgcn_wmma_f32_16x16x32_f16(false, aq0, false, b10, (short)0, s1, false, false);
      s1 = __builtin_amdgcn_wmma_f32_16x16x32_f16(false, aq1, false, b11, (short)0, s1, false, false);
    }
    // causal mask (reference adds -10000) + online softmax
    float p0[8], p1[8];
#pragma unroll
    for (int r = 0; r < 8; ++r) {
      int row = qr + r + (hs << 3);
      if (kc + m > row)      s0[r] += -10000.0f;
      if (kc + 16 + m > row) s1[r] += -10000.0f;
      float t = fmaxf(s0[r], s1[r]);
#pragma unroll
      for (int off = 1; off < 16; off <<= 1) t = fmaxf(t, __shfl_xor(t, off, 32));
      float mn = fmaxf(mrow[r], t);
      float fr = __expf(mrow[r] - mn);
      p0[r] = __expf(s0[r] - mn);
      p1[r] = __expf(s1[r] - mn);
      float rs = p0[r] + p1[r];
#pragma unroll
      for (int off = 1; off < 16; off <<= 1) rs += __shfl_xor(rs, off, 32);
      lrow[r] = lrow[r] * fr + rs;
      mrow[r] = mn;
      oacc[0][r] *= fr; oacc[1][r] *= fr; oacc[2][r] *= fr; oacc[3][r] *= fr;
    }
    // C-layout -> A-layout bounce through per-wave LDS slice
#pragma unroll
    for (int r = 0; r < 8; ++r) {
      int row = r + (hs << 3);
      pls[wv][row][m]      = (_Float16)p0[r];
      pls[wv][row][16 + m] = (_Float16)p1[r];
    }
    __syncthreads();
    v16h ap = ld16(&pls[wv][m][hs * 8], &pls[wv][m][16 + hs * 8]);
    __syncthreads();
    // P(16x32) @ V(32x64); V is stored transposed -> contiguous B-frags
#pragma unroll
    for (int dt = 0; dt < 4; ++dt) {
      v16h bv = ldB(vT + (size_t)(dt * 16 + m) * L_, kc, hs);
      oacc[dt] = __builtin_amdgcn_wmma_f32_16x16x32_f16(false, ap, false, bv, (short)0, oacc[dt], false, false);
    }
  }
#pragma unroll
  for (int r = 0; r < 8; ++r) {
    int row = qr + r + (hs << 3);
    float inv = 1.0f / lrow[r];
#pragma unroll
    for (int dt = 0; dt < 4; ++dt)
      ysm[((size_t)bh * L_ + row) * DH + dt * 16 + m] = oacc[dt][r] * inv;
  }
}

// ---------------------------------------------------------------------------
// alpha = sigmoid(h @ gate_w2 + b2); tiny GEMM, one thread per (token, head)
// ---------------------------------------------------------------------------
__global__ void alpha_kernel(const float* __restrict__ hb, const float* __restrict__ w2,
                             const float* __restrict__ b2, float* __restrict__ alpha) {
  int id = blockIdx.x * blockDim.x + threadIdx.x;       // 65536
  int t = id >> 4, hh = id & 15;
  float acc = b2[hh];
  for (int j = 0; j < 512; ++j)
    acc += hb[(size_t)t * 512 + j] * w2[j * 16 + hh];
  alpha[id] = sigmoidf_(acc);
}

// ---------------------------------------------------------------------------
// STP phase A: per-chunk suffix state (64x64 state in 16 regs/thread).
// ---------------------------------------------------------------------------
__global__ void stp_phaseA(const float* __restrict__ qkv32, const float* __restrict__ Lambda,
                           const float* __restrict__ Gamma, float* __restrict__ Achunk) {
  const int c = blockIdx.x % NC, bh = blockIdx.x / NC, hh = bh % H_;
  const int tid = threadIdx.x;
  const int i = tid & 63, jb = (tid >> 6) << 4;
  const float* kptr = qkv32 + ((size_t)(1 * BH + bh) * L_) * DH;
  const float* vptr = qkv32 + ((size_t)(2 * BH + bh) * L_) * DH;
  float ret[16], gam[16], st[16];
#pragma unroll
  for (int e = 0; e < 16; ++e) {
    size_t idx = (size_t)hh * 4096 + (size_t)i * 64 + jb + e;
    ret[e] = 1.0f - sigmoidf_(Lambda[idx]);
    gam[e] = Gamma[idx];
    st[e]  = 0.0f;
  }
  __shared__ float kb_[64], vb_[64];
  for (int t = c * CH; t < (c + 1) * CH; ++t) {
    __syncthreads();
    if (tid < 64)       kb_[tid]      = kptr[(size_t)t * DH + tid];
    else if (tid < 128) vb_[tid - 64] = vptr[(size_t)t * DH + (tid - 64)];
    __syncthreads();
    float vi = vb_[i];
#pragma unroll
    for (int e = 0; e < 16; ++e)
      st[e] = ret[e] * st[e] + gam[e] * vi * kb_[jb + e];
  }
  size_t base = ((size_t)bh * NC + c) * 4096 + (size_t)i * 64 + jb;
#pragma unroll
  for (int e = 0; e < 16; ++e) Achunk[base + e] = st[e];
}

// ---------------------------------------------------------------------------
// STP phase B: 16-step checkpoint chain  E_{c+1} = ret^CH * E_c + A_c
// ---------------------------------------------------------------------------
__global__ void stp_scan(const float* __restrict__ Lambda, const float* __restrict__ Achunk,
                         float* __restrict__ Eckpt) {
  const int bh = blockIdx.x, hh = bh % H_;
  const int tid = threadIdx.x;
  const int i = tid & 63, jb = (tid >> 6) << 4;
  float retC[16], E[16];
#pragma unroll
  for (int e = 0; e < 16; ++e) {
    float r = 1.0f - sigmoidf_(Lambda[(size_t)hh * 4096 + (size_t)i * 64 + jb + e]);
    retC[e] = powf(r, (float)CH);
    E[e] = 0.0f;
  }
  for (int c = 0; c < NC; ++c) {
    size_t base = ((size_t)bh * NC + c) * 4096 + (size_t)i * 64 + jb;
#pragma unroll
    for (int e = 0; e < 16; ++e) {
      Eckpt[base + e] = E[e];                 // state entering chunk c
      E[e] = retC[e] * E[e] + Achunk[base + e];
    }
  }
}

// ---------------------------------------------------------------------------
// STP phase C: replay chunk from checkpoint, y_t = (W_static + state_t) @ q_t
// ---------------------------------------------------------------------------
__global__ void stp_phaseC(const float* __restrict__ qkv32, const float* __restrict__ Lambda,
                           const float* __restrict__ Gamma, const float* __restrict__ Wst,
                           const float* __restrict__ Eckpt, float* __restrict__ ystp) {
  const int c = blockIdx.x % NC, bh = blockIdx.x / NC, hh = bh % H_;
  const int tid = threadIdx.x;
  const int i = tid & 63, jg = tid >> 6, jb = jg << 4;
  const float* qptr = qkv32 + ((size_t)(0 * BH + bh) * L_) * DH;
  const float* kptr = qkv32 + ((size_t)(1 * BH + bh) * L_) * DH;
  const float* vptr = qkv32 + ((size_t)(2 * BH + bh) * L_) * DH;
  float ret[16], gam[16], st[16], ws[16];
  size_t ebase = ((size_t)bh * NC + c) * 4096 + (size_t)i * 64 + jb;
#pragma unroll
  for (int e = 0; e < 16; ++e) {
    size_t idx = (size_t)hh * 4096 + (size_t)i * 64 + jb + e;
    ret[e] = 1.0f - sigmoidf_(Lambda[idx]);
    gam[e] = Gamma[idx];
    ws[e]  = Wst[idx];
    st[e]  = Eckpt[ebase + e];
  }
  __shared__ float kb_[64], vb_[64], qb_[64], part[256];
  for (int t = c * CH; t < (c + 1) * CH; ++t) {
    __syncthreads();
    if (tid < 64)       kb_[tid]       = kptr[(size_t)t * DH + tid];
    else if (tid < 128) vb_[tid - 64]  = vptr[(size_t)t * DH + (tid - 64)];
    else if (tid < 192) qb_[tid - 128] = qptr[(size_t)t * DH + (tid - 128)];
    __syncthreads();
    float vi = vb_[i], acc = 0.0f;
#pragma unroll
    for (int e = 0; e < 16; ++e) {
      st[e] = ret[e] * st[e] + gam[e] * vi * kb_[jb + e];
      acc += (ws[e] + st[e]) * qb_[jb + e];
    }
    part[(jg << 6) + i] = acc;
    __syncthreads();
    if (tid < 64)
      ystp[((size_t)bh * L_ + t) * DH + i] =
          part[i] + part[64 + i] + part[128 + i] + part[192 + i];
  }
}

// ---------------------------------------------------------------------------
// y = alpha*y_softmax + (1-alpha)*y_stp, relayout (B,H,L,d) -> (B,L,D), f16 out
// ---------------------------------------------------------------------------
__global__ void combine_kernel(const float* __restrict__ ysm, const float* __restrict__ ystp,
                               const float* __restrict__ alpha, _Float16* __restrict__ yc16) {
  size_t id = (size_t)blockIdx.x * blockDim.x + threadIdx.x;   // over BH*L*64
  int feat = (int)(id & 63);
  size_t rest = id >> 6;
  int l  = (int)(rest % L_);
  int bh = (int)(rest / L_);
  int b = bh / H_, hh = bh % H_;
  float a = alpha[((size_t)b * L_ + l) * H_ + hh];
  yc16[((size_t)b * L_ + l) * D_ + hh * DH + feat] =
      (_Float16)(a * ysm[id] + (1.0f - a) * ystp[id]);
}

extern "C" void kernel_launch(void* const* d_in, const int* in_sizes, int n_in,
                              void* d_out, int out_size, void* d_ws, size_t ws_size,
                              hipStream_t stream) {
  (void)in_sizes; (void)n_in; (void)out_size; (void)ws_size;
  const float* x        = (const float*)d_in[0];
  const float* Wqkv_w   = (const float*)d_in[1];
  const float* Wqkv_b   = (const float*)d_in[2];
  const float* out_w    = (const float*)d_in[3];
  const float* out_b    = (const float*)d_in[4];
  const float* W_static = (const float*)d_in[5];
  const float* Lambda   = (const float*)d_in[6];
  const float* Gamma    = (const float*)d_in[7];
  const float* gate_w1  = (const float*)d_in[8];
  const float* gate_b1  = (const float*)d_in[9];
  const float* gate_w2  = (const float*)d_in[10];
  const float* gate_b2  = (const float*)d_in[11];

  char* ws = (char*)d_ws;
  size_t off = 0;
  auto alloc = [&](size_t bytes) -> char* {
    off = (off + 255) & ~(size_t)255;
    char* p = ws + off;
    off += bytes;
    return p;
  };
  const size_t PK = (size_t)3 * BH * L_ * DH;                  // 12.58M elems
  float*    qkv32  = (float*)alloc(PK * 4);                    // ~50 MB
  _Float16* qkv16  = (_Float16*)alloc(PK * 2);                 // ~25 MB
  _Float16* x16    = (_Float16*)alloc((size_t)4096 * 1024 * 2);
  _Float16* WqkvT  = (_Float16*)alloc((size_t)3072 * 1024 * 2);
  _Float16* gw1T   = (_Float16*)alloc((size_t)512 * 1024 * 2);
  _Float16* outwT  = (_Float16*)alloc((size_t)1024 * 1024 * 2);
  float*    hbuf   = (float*)alloc((size_t)4096 * 512 * 4);
  float*    alpha  = (float*)alloc((size_t)4096 * 16 * 4);
  float*    ysm    = (float*)alloc((size_t)BH * L_ * DH * 4);
  float*    ystp   = (float*)alloc((size_t)BH * L_ * DH * 4);
  _Float16* yc16   = (_Float16*)alloc((size_t)4096 * 1024 * 2);
  float*    Achunk = (float*)alloc((size_t)BH * NC * 4096 * 4);
  float*    Eckpt  = (float*)alloc((size_t)BH * NC * 4096 * 4);

  // 0) pack operands to f16 (weights transposed for contiguous B-frags)
  cvt_f16<<<dim3(16384), dim3(256), 0, stream>>>(x, x16, 4096 * 1024);
  transpose_cvt<<<dim3(12288), dim3(256), 0, stream>>>(Wqkv_w, WqkvT, 1024, 3072);
  transpose_cvt<<<dim3(2048),  dim3(256), 0, stream>>>(gate_w1, gw1T, 1024, 512);
  transpose_cvt<<<dim3(4096),  dim3(256), 0, stream>>>(out_w, outwT, 1024, 1024);

  // 1) QKV projection + pack (f32 row-major for STP; f16 with v transposed)
  qkv_gemm<<<dim3(1536), dim3(256), 0, stream>>>(x16, WqkvT, Wqkv_b, qkv32, qkv16);
  // 2) gate hidden: gelu(x @ gate_w1 + b1)
  gemm_f16wmma<<<dim3(256), dim3(256), 0, stream>>>(x16, gw1T, gate_b1, hbuf,
                                                    4096, 512, 1024, 1);
  // 3) alpha = sigmoid(h @ gate_w2 + b2)
  alpha_kernel<<<dim3(256), dim3(256), 0, stream>>>(hbuf, gate_w2, gate_b2, alpha);
  // 4) causal flash attention
  attn_kernel<<<dim3(BH * (L_ / 64)), dim3(128), 0, stream>>>(qkv16, ysm);
  // 5-7) STP chunked scan
  stp_phaseA<<<dim3(BH * NC), dim3(256), 0, stream>>>(qkv32, Lambda, Gamma, Achunk);
  stp_scan  <<<dim3(BH),      dim3(256), 0, stream>>>(Lambda, Achunk, Eckpt);
  stp_phaseC<<<dim3(BH * NC), dim3(256), 0, stream>>>(qkv32, Lambda, Gamma, W_static,
                                                      Eckpt, ystp);
  // 8) gate-combine + relayout (f16 for final GEMM A operand)
  combine_kernel<<<dim3((BH * L_ * DH) / 256), dim3(256), 0, stream>>>(ysm, ystp, alpha, yc16);
  // 9) output projection
  gemm_f16wmma<<<dim3(512), dim3(256), 0, stream>>>(yc16, outwT, out_b, (float*)d_out,
                                                    4096, 1024, 1024, 0);
}
